// Decoder_SelfAttn_51041391346064
// MI455X (gfx1250) — matrix operand order repaired and checked by reading, not compiled, not run on hardware
//
#include <hip/hip_runtime.h>
#include <hip/hip_bf16.h>
#include <math.h>

#define HID    1024
#define H3     3072
#define VOC    32000
#define MEMN   32
#define BATCH  32
#define TSTEPS 64

typedef __attribute__((ext_vector_type(16))) __bf16 v16bf;
typedef __attribute__((ext_vector_type(8)))  __bf16 v8bf;
typedef __attribute__((ext_vector_type(8)))  float  v8f;

union V16U { v16bf v; v8bf h[2]; };

__device__ __forceinline__ v8f wmma_bf16(v16bf a, v16bf b, v8f c) {
  // D = A(16x32) x B(32x16) + C, f32 accumulate
  return __builtin_amdgcn_wmma_f32_16x16x32_bf16(false, a, false, b, (short)0, c,
                                                 false, false);
}

// ---------------------------------------------------------------- utilities
__global__ void convert_bf16_kernel(const float* __restrict__ src,
                                    __bf16* __restrict__ dst, size_t n) {
  size_t i = (size_t)blockIdx.x * blockDim.x + threadIdx.x;
  if (i < n) dst[i] = (__bf16)src[i];
}

// W_proj (H, 2H) -> Wp1 (H,H) [x half], Wp2 (H,H) [mem half], bf16
__global__ void split_wproj_kernel(const float* __restrict__ wp,
                                   __bf16* __restrict__ wp1,
                                   __bf16* __restrict__ wp2) {
  size_t i = (size_t)blockIdx.x * blockDim.x + threadIdx.x;  // over H*H
  size_t n = i / HID, k = i % HID;
  wp1[i] = (__bf16)wp[n * (2 * HID) + k];
  wp2[i] = (__bf16)wp[n * (2 * HID) + HID + k];
}

// X[t*B+b][h] = bf16(relu(emb_table[ids[b][t]][h]))
__global__ void embed_relu_kernel(const int* __restrict__ ids,
                                  const float* __restrict__ table,
                                  __bf16* __restrict__ X) {
  int bt = blockIdx.x;                 // = t*BATCH + b
  int t = bt / BATCH, b = bt % BATCH;
  int id = ids[b * TSTEPS + t];
  const float* src = table + (size_t)id * HID;
  __bf16* dst = X + (size_t)bt * HID;
  for (int h = threadIdx.x; h < HID; h += blockDim.x) {
    float v = src[h];
    dst[h] = (__bf16)(v > 0.f ? v : 0.f);
  }
}

// Ring init: R[b][m][h] = memory[m][b][h]; hfb = bf16(memory[0][b][:])
__global__ void init_ring_kernel(const float* __restrict__ memory,
                                 float* __restrict__ R,
                                 __bf16* __restrict__ hfb) {
  size_t i = (size_t)blockIdx.x * blockDim.x + threadIdx.x;  // over MEM*B*H
  size_t h = i % HID, rest = i / HID;
  size_t b = rest % BATCH, m = rest / BATCH;
  float v = memory[(m * BATCH + b) * HID + h];
  R[((size_t)b * MEMN + m) * HID + h] = v;
  if (m == 0) hfb[b * HID + h] = (__bf16)v;
}

// -------------------------------------------------------------- GEMM kernels
// C[M,N] = A[M,K] @ W[N,K]^T (+bias) (+Cin). One 16x16 tile per wave.
__global__ void gemm16_kernel(const __bf16* __restrict__ A, int lda,
                              const __bf16* __restrict__ W, int ldw,
                              const float* __restrict__ bias,
                              const float* __restrict__ Cin, int ldcin,
                              float* __restrict__ C, int ldc,
                              int M, int N, int K) {
  int wave = (blockIdx.x * blockDim.x + threadIdx.x) >> 5;
  int ntn = N >> 4;
  int m0 = (wave / ntn) << 4;
  int n0 = (wave % ntn) << 4;
  if (m0 >= M) return;
  int lane = threadIdx.x & 31, half = lane >> 4, r = lane & 15;
  const __bf16* ap = A + (size_t)(m0 + r) * lda + half * 16;
  const __bf16* wp = W + (size_t)(n0 + r) * ldw + half * 16;
  v8f acc = {};
  for (int k = 0; k < K; k += 32) {
    V16U a;
    a.h[0] = *(const v8bf*)(ap + k);
    a.h[1] = *(const v8bf*)(ap + k + 8);
    v16bf bfr = *(const v16bf*)(wp + k);
    acc = wmma_bf16(a.v, bfr, acc);
  }
  int col = n0 + r;
  float bv = bias ? bias[col] : 0.f;
#pragma unroll
  for (int e = 0; e < 8; ++e) {
    int row = m0 + e + 8 * half;
    float v = acc[e] + bv;
    if (Cin) v += Cin[(size_t)row * ldcin + col];
    C[(size_t)row * ldc + col] = v;
  }
}

// 16x64 tile per wave: A fragment reused across 4 WMMAs (for big GEMMs).
__global__ void gemm64_kernel(const __bf16* __restrict__ A, int lda,
                              const __bf16* __restrict__ W, int ldw,
                              const float* __restrict__ bias,
                              float* __restrict__ C, int ldc,
                              int M, int N, int K) {
  int wave = (blockIdx.x * blockDim.x + threadIdx.x) >> 5;
  int ntn = N >> 6;
  int m0 = (wave / ntn) << 4;
  int n0 = (wave % ntn) << 6;
  if (m0 >= M) return;
  int lane = threadIdx.x & 31, half = lane >> 4, r = lane & 15;
  const __bf16* ap = A + (size_t)(m0 + r) * lda + half * 16;
  const __bf16* wp = W + (size_t)(n0 + r) * ldw + half * 16;
  size_t wstep = (size_t)16 * ldw;
  v8f a0 = {}, a1 = {}, a2 = {}, a3 = {};
  for (int k = 0; k < K; k += 32) {
    V16U a;
    a.h[0] = *(const v8bf*)(ap + k);
    a.h[1] = *(const v8bf*)(ap + k + 8);
    v16bf b0 = *(const v16bf*)(wp + k);
    v16bf b1 = *(const v16bf*)(wp + wstep + k);
    v16bf b2 = *(const v16bf*)(wp + 2 * wstep + k);
    v16bf b3 = *(const v16bf*)(wp + 3 * wstep + k);
    a0 = wmma_bf16(a.v, b0, a0);
    a1 = wmma_bf16(a.v, b1, a1);
    a2 = wmma_bf16(a.v, b2, a2);
    a3 = wmma_bf16(a.v, b3, a3);
  }
  v8f accs[4] = {a0, a1, a2, a3};
#pragma unroll
  for (int j = 0; j < 4; ++j) {
    int col = n0 + 16 * j + r;
    float bv = bias ? bias[col] : 0.f;
#pragma unroll
    for (int e = 0; e < 8; ++e) {
      int row = m0 + e + 8 * half;
      C[(size_t)row * ldc + col] = accs[j][e] + bv;
    }
  }
}

// ------------------------------------------------------------ recurrent step
// dots = (mem . proj)/sqrt(H); softmax; sel = w @ mem. One block per batch.
__global__ void attn_kernel(const float* __restrict__ R,
                            const float* __restrict__ proj,
                            float* __restrict__ sel,
                            __bf16* __restrict__ selb, int p) {
  int b = blockIdx.x, tid = threadIdx.x, lane = tid & 31, wv = tid >> 5;
  __shared__ float wgt[MEMN];
  const float* pr = proj + (size_t)b * HID;
#pragma unroll
  for (int i = 0; i < 4; ++i) {
    int m = wv * 4 + i;
    int phys = (p + m) & (MEMN - 1);
    const float* row = R + ((size_t)b * MEMN + phys) * HID;
    float s = 0.f;
    for (int j = lane; j < HID; j += 32) s += row[j] * pr[j];
    for (int o = 16; o; o >>= 1) s += __shfl_xor(s, o);
    if (lane == 0) wgt[m] = s * 0.03125f;  // 1/sqrt(1024)
  }
  __syncthreads();
  if (tid < 32) {
    float v = wgt[tid];
    float mx = v;
    for (int o = 16; o; o >>= 1) mx = fmaxf(mx, __shfl_xor(mx, o));
    float e = expf(v - mx);
    float sum = e;
    for (int o = 16; o; o >>= 1) sum += __shfl_xor(sum, o);
    wgt[tid] = e / sum;
  }
  __syncthreads();
  for (int h = tid; h < HID; h += blockDim.x) {
    float acc = 0.f;
#pragma unroll 8
    for (int m = 0; m < MEMN; ++m) {
      int phys = (p + m) & (MEMN - 1);
      acc += wgt[m] * R[((size_t)b * MEMN + phys) * HID + h];
    }
    sel[(size_t)b * HID + h] = acc;
    selb[(size_t)b * HID + h] = (__bf16)acc;
  }
}

// GRU gates; writes h_new into ring front (pw) and bf16 copy for GEMMs.
__global__ void gru_kernel(const float* __restrict__ gi,
                           const float* __restrict__ gh,
                           const float* __restrict__ sel,
                           float* __restrict__ R,
                           __bf16* __restrict__ hfb, int pw) {
  int i = blockIdx.x * blockDim.x + threadIdx.x;  // over B*H
  int b = i >> 10, h = i & (HID - 1);
  size_t gbase = (size_t)b * H3 + h;
  float i_r = gi[gbase], i_z = gi[gbase + HID], i_n = gi[gbase + 2 * HID];
  float h_r = gh[gbase], h_z = gh[gbase + HID], h_n = gh[gbase + 2 * HID];
  float r = 1.f / (1.f + expf(-(i_r + h_r)));
  float z = 1.f / (1.f + expf(-(i_z + h_z)));
  float n = tanhf(i_n + r * h_n);
  float hv = sel[(size_t)b * HID + h];
  float hn = (1.f - z) * n + z * hv;
  R[((size_t)b * MEMN + pw) * HID + h] = hn;
  hfb[(size_t)b * HID + h] = (__bf16)hn;
}

// Row-wise log_softmax over V=32000; one block (8 waves) per batch row.
__global__ void logsoftmax_kernel(const float* __restrict__ logits,
                                  float* __restrict__ out, int t) {
  int b = blockIdx.x, tid = threadIdx.x, lane = tid & 31, wv = tid >> 5;
  const float* row = logits + (size_t)b * VOC;
  __shared__ float red[8];
  float mx = -3.402823466e38f;
  for (int v = tid; v < VOC; v += blockDim.x) mx = fmaxf(mx, row[v]);
  for (int o = 16; o; o >>= 1) mx = fmaxf(mx, __shfl_xor(mx, o));
  if (lane == 0) red[wv] = mx;
  __syncthreads();
  if (tid == 0) {
    float m = red[0];
    for (int i = 1; i < 8; ++i) m = fmaxf(m, red[i]);
    red[0] = m;
  }
  __syncthreads();
  mx = red[0];
  __syncthreads();
  float s = 0.f;
  for (int v = tid; v < VOC; v += blockDim.x) s += expf(row[v] - mx);
  for (int o = 16; o; o >>= 1) s += __shfl_xor(s, o);
  if (lane == 0) red[wv] = s;
  __syncthreads();
  if (tid == 0) {
    float tot = 0.f;
    for (int i = 1; i < 8; ++i) tot += red[i];
    red[0] = logf(tot + red[0]);
  }
  __syncthreads();
  float ls = red[0];
  float* o = out + ((size_t)b * TSTEPS + t) * VOC;
  for (int v = tid; v < VOC; v += blockDim.x) o[v] = row[v] - mx - ls;
}

// After T=64 steps p wraps to 0: ring is identity-mapped -> direct copy.
__global__ void copy_mem_kernel(const float* __restrict__ R,
                                float* __restrict__ out) {
  size_t i = (size_t)blockIdx.x * blockDim.x + threadIdx.x;
  out[i] = R[i];
}

// --------------------------------------------------------------------- host
extern "C" void kernel_launch(void* const* d_in, const int* in_sizes, int n_in,
                              void* d_out, int out_size, void* d_ws, size_t ws_size,
                              hipStream_t stream) {
  const int*   ids     = (const int*)d_in[0];
  const float* memory  = (const float*)d_in[1];
  const float* emb_tab = (const float*)d_in[2];
  const float* W_proj  = (const float*)d_in[3];
  // b_proj = d_in[4]
  const float* b_proj  = (const float*)d_in[4];
  const float* W_ih    = (const float*)d_in[5];
  const float* W_hh    = (const float*)d_in[6];
  const float* b_ih    = (const float*)d_in[7];
  const float* b_hh    = (const float*)d_in[8];
  const float* W_out   = (const float*)d_in[9];
  const float* b_out   = (const float*)d_in[10];

  float* out_scores = (float*)d_out;
  float* out_mem    = out_scores + (size_t)BATCH * TSTEPS * VOC;

  char* ws = (char*)d_ws;
  size_t off = 0;
  auto alloc = [&](size_t bytes) -> void* {
    void* p = ws + off;
    off = (off + bytes + 255) & ~(size_t)255;
    return p;
  };
  __bf16* Woutb = (__bf16*)alloc((size_t)VOC * HID * 2);
  __bf16* Wihb  = (__bf16*)alloc((size_t)H3 * HID * 2);
  __bf16* Whhb  = (__bf16*)alloc((size_t)H3 * HID * 2);
  __bf16* Wp1b  = (__bf16*)alloc((size_t)HID * HID * 2);
  __bf16* Wp2b  = (__bf16*)alloc((size_t)HID * HID * 2);
  __bf16* Xb    = (__bf16*)alloc((size_t)TSTEPS * BATCH * HID * 2);
  float*  GI    = (float*) alloc((size_t)TSTEPS * BATCH * H3 * 4);
  float*  PX    = (float*) alloc((size_t)TSTEPS * BATCH * HID * 4);
  float*  R     = (float*) alloc((size_t)BATCH * MEMN * HID * 4);
  __bf16* hfb   = (__bf16*)alloc((size_t)BATCH * HID * 2);
  float*  proj  = (float*) alloc((size_t)BATCH * HID * 4);
  float*  sel   = (float*) alloc((size_t)BATCH * HID * 4);
  __bf16* selb  = (__bf16*)alloc((size_t)BATCH * HID * 2);
  float*  gh    = (float*) alloc((size_t)BATCH * H3 * 4);
  float*  lg    = (float*) alloc((size_t)BATCH * VOC * 4);

  const int TB = 256;
  auto cgrid = [](size_t n, int tb) { return (int)((n + tb - 1) / tb); };

  // ---- one-time prep: weight conversion (bf16), embeddings, ring init
  convert_bf16_kernel<<<cgrid((size_t)VOC * HID, TB), TB, 0, stream>>>(W_out, Woutb, (size_t)VOC * HID);
  convert_bf16_kernel<<<cgrid((size_t)H3 * HID, TB), TB, 0, stream>>>(W_ih, Wihb, (size_t)H3 * HID);
  convert_bf16_kernel<<<cgrid((size_t)H3 * HID, TB), TB, 0, stream>>>(W_hh, Whhb, (size_t)H3 * HID);
  split_wproj_kernel<<<cgrid((size_t)HID * HID, TB), TB, 0, stream>>>(W_proj, Wp1b, Wp2b);
  embed_relu_kernel<<<TSTEPS * BATCH, TB, 0, stream>>>(ids, emb_tab, Xb);
  init_ring_kernel<<<cgrid((size_t)MEMN * BATCH * HID, TB), TB, 0, stream>>>(memory, R, hfb);

  // ---- hoisted big GEMMs: GI = X@W_ih^T + b_ih ; PX = X@Wp1^T + b_proj
  {
    int waves = (TSTEPS * BATCH / 16) * (H3 / 64);
    gemm64_kernel<<<waves * 32 / TB, TB, 0, stream>>>(Xb, HID, Wihb, HID, b_ih,
                                                      GI, H3, TSTEPS * BATCH, H3, HID);
  }
  {
    int waves = (TSTEPS * BATCH / 16) * (HID / 64);
    gemm64_kernel<<<waves * 32 / TB, TB, 0, stream>>>(Xb, HID, Wp1b, HID, b_proj,
                                                      PX, HID, TSTEPS * BATCH, HID, HID);
  }

  // ---- serialized decode loop
  for (int t = 0; t < TSTEPS; ++t) {
    int p  = (MEMN - (t % MEMN)) % MEMN;   // logical row 0 -> physical p
    int pw = (p + MEMN - 1) % MEMN;        // where h_new lands

    // proj = PX[t] + h_front @ Wp2^T
    {
      int waves = (BATCH / 16) * (HID / 16);
      gemm16_kernel<<<waves * 32 / TB, TB, 0, stream>>>(
          hfb, HID, Wp2b, HID, nullptr,
          PX + (size_t)t * BATCH * HID, HID, proj, HID, BATCH, HID, HID);
    }
    attn_kernel<<<BATCH, TB, 0, stream>>>(R, proj, sel, selb, p);
    // gh = sel @ W_hh^T + b_hh
    {
      int waves = (BATCH / 16) * (H3 / 16);
      gemm16_kernel<<<waves * 32 / TB, TB, 0, stream>>>(
          selb, HID, Whhb, HID, b_hh, nullptr, 0, gh, H3, BATCH, H3, HID);
    }
    gru_kernel<<<(BATCH * HID) / TB, TB, 0, stream>>>(
        GI + (size_t)t * BATCH * H3, gh, sel, R, hfb, pw);
    // logits = h_new @ W_out^T + b_out   (W_out bf16 is L2-resident: 65.5MB)
    {
      int waves = (BATCH / 16) * (VOC / 64);
      gemm64_kernel<<<waves * 32 / TB, TB, 0, stream>>>(
          hfb, HID, Woutb, HID, b_out, lg, VOC, BATCH, VOC, HID);
    }
    logsoftmax_kernel<<<BATCH, TB, 0, stream>>>(lg, out_scores, t);
  }

  copy_mem_kernel<<<(BATCH * MEMN * HID) / TB, TB, 0, stream>>>(R, out_mem);
}
